// Normalized_Correlation_Layer_12816182411489
// MI455X (gfx1250) — matrix operand order repaired and checked by reading, not compiled, not run on hardware
//
#include <hip/hip_runtime.h>
#include <hip/hip_bf16.h>

typedef __attribute__((ext_vector_type(16))) _Float16 v16h;
typedef __attribute__((ext_vector_type(8)))  float    v8f;
typedef __attribute__((ext_vector_type(8)))  int      v8i;

#define BB 8
#define HH 32
#define WW 32
#define CC 64
// out[b,r,w,k,e,f] : stride of w is CC*5*WW = 10240 floats
#define WSTRIDE (CC * 5 * WW)

__global__ __launch_bounds__(256) void ncl_wmma_kernel(
    const float* __restrict__ x1, const float* __restrict__ x2,
    float* __restrict__ out)
{
  // LDS staging with zero halo: [channel][row][padded col]
  __shared__ float x1s[8][5][36];   // rows y = r-2 .. r+2, cols x = -2 .. 33
  __shared__ float x2s[8][9][36];   // rows y = r-4 .. r+4, cols x = -2 .. 33

  const int tid = threadIdx.x;
  // block -> (b, r, 8-channel group)
  const int kg = blockIdx.x & 7;
  const int r  = (blockIdx.x >> 3) & (HH - 1);
  const int b  = blockIdx.x >> 8;
  const int kb = kg * 8;

  const float* x1b = x1 + (size_t)(b * HH) * WW * CC + kb;
  const float* x2b = x2 + (size_t)(b * HH) * WW * CC + kb;

  // ---- cooperative fill, branchless (clamped address + select) ----
  for (int f = tid; f < 8 * 5 * 36; f += 256) {
    const int kk  = f & 7;
    const int cr  = f >> 3;
    const int col = cr % 36;
    const int row = cr / 36;
    const int y = r + row - 2, x = col - 2;
    const int yc = min(max(y, 0), HH - 1), xc = min(max(x, 0), WW - 1);
    const float v = x1b[(yc * WW + xc) * CC + kk];       // always in-range
    const bool ok = ((unsigned)y < (unsigned)HH) && ((unsigned)x < (unsigned)WW);
    x1s[kk][row][col] = ok ? v : 0.f;
  }
  for (int f = tid; f < 8 * 9 * 36; f += 256) {
    const int kk  = f & 7;
    const int cr  = f >> 3;
    const int col = cr % 36;
    const int row = cr / 36;
    const int y = r + row - 4, x = col - 2;
    const int yc = min(max(y, 0), HH - 1), xc = min(max(x, 0), WW - 1);
    const float v = x2b[(yc * WW + xc) * CC + kk];
    const bool ok = ((unsigned)y < (unsigned)HH) && ((unsigned)x < (unsigned)WW);
    x2s[kk][row][col] = ok ? v : 0.f;
  }
  __syncthreads();

  // ---- per-wave GEMM: wave kk handles channel k = kb + kk ----
  const int kk   = tid >> 5;
  const int lane = tid & 31;
  const int half = lane >> 4;   // 0 = lanes 0-15, 1 = lanes 16-31
  const int l    = lane & 15;
  const int k    = kb + kk;

  float patch[25];

  // =============== A fragments: each lane owns patch w = lane ===============
  v16h afrag[2];
  {
    float s0 = 0.f, s1 = 0.f, s2 = 0.f, s3 = 0.f;
    float q0 = 0.f, q1 = 0.f, q2 = 0.f, q3 = 0.f;
#pragma unroll
    for (int i = 0; i < 5; ++i)
#pragma unroll
      for (int j = 0; j < 5; ++j) {
        const int p = i * 5 + j;
        const float v = x1s[kk][i][lane + j];   // w = lane, halo zeroed
        patch[p] = v;
        switch (p & 3) {
          case 0: s0 += v; q0 = fmaf(v, v, q0); break;
          case 1: s1 += v; q1 = fmaf(v, v, q1); break;
          case 2: s2 += v; q2 = fmaf(v, v, q2); break;
          default: s3 += v; q3 = fmaf(v, v, q3); break;
        }
      }
    const float m   = ((s0 + s1) + (s2 + s3)) * 0.04f;
    const float ex2 = ((q0 + q1) + (q2 + q3)) * 0.04f;
    const float inv  = __frsqrt_rn(fmaf(-m, m, ex2));
    const float bias = -m * inv;

    // al: K 0-7,16-23 ; ah: K 8-15,24 (rest zero)  [16-bit A 16x32 layout]
    v16h al, ah;
#pragma unroll
    for (int h = 0; h < 8; ++h) {
      al[h]     = (_Float16)fmaf(patch[h],      inv, bias);  // K 0-7
      al[8 + h] = (_Float16)fmaf(patch[16 + h], inv, bias);  // K 16-23
      ah[h]     = (_Float16)fmaf(patch[8 + h],  inv, bias);  // K 8-15
    }
    ah[8] = (_Float16)fmaf(patch[24], inv, bias);            // K 24
#pragma unroll
    for (int h = 9; h < 16; ++h) ah[h] = (_Float16)0.f;      // K 25-31

    const v8i ali = __builtin_bit_cast(v8i, al);
    const v8i ahi = __builtin_bit_cast(v8i, ah);
    v8i alx, ahx;
#pragma unroll
    for (int t = 0; t < 8; ++t) alx[t] = __shfl_xor(ali[t], 16, 32);
#pragma unroll
    for (int t = 0; t < 5; ++t) ahx[t] = __shfl_xor(ahi[t], 16, 32);
#pragma unroll
    for (int t = 5; t < 8; ++t) ahx[t] = 0;

    v8i f0, f1;
#pragma unroll
    for (int t = 0; t < 8; ++t) {
      f0[t] = half ? ahx[t] : ali[t];   // rows 0-15
      f1[t] = half ? ahi[t] : alx[t];   // rows 16-31
    }
    afrag[0] = __builtin_bit_cast(v16h, f0);
    afrag[1] = __builtin_bit_cast(v16h, f1);
  }

  const int obase = ((b * HH + r) * WW * CC + k) * (5 * WW);

  // =============== B side: per e, each lane owns patch f = lane =============
#pragma unroll
  for (int e = 0; e < 5; ++e) {
    // rows_idx mapping: q = (r+e <= 33) ? r+e : r+e-2 ; LDS row = (q-r) + i
    const int qmr = (r + e <= 33) ? e : e - 2;

    float s0 = 0.f, s1 = 0.f, s2 = 0.f, s3 = 0.f;
    float q0 = 0.f, q1 = 0.f, q2 = 0.f, q3 = 0.f;
#pragma unroll
    for (int i = 0; i < 5; ++i)
#pragma unroll
      for (int j = 0; j < 5; ++j) {
        const int p = i * 5 + j;
        const float v = x2s[kk][qmr + i][lane + j];   // f = lane
        patch[p] = v;
        switch (p & 3) {
          case 0: s0 += v; q0 = fmaf(v, v, q0); break;
          case 1: s1 += v; q1 = fmaf(v, v, q1); break;
          case 2: s2 += v; q2 = fmaf(v, v, q2); break;
          default: s3 += v; q3 = fmaf(v, v, q3); break;
        }
      }
    const float m   = ((s0 + s1) + (s2 + s3)) * 0.04f;
    const float ex2 = ((q0 + q1) + (q2 + q3)) * 0.04f;
    const float inv  = __frsqrt_rn(fmaf(-m, m, ex2));
    const float bias = -m * inv;

    // bl: K 0-7,16-23 ; bh: K 8-15,24 (rest zero)   [16-bit B 32x16 layout]
    v16h bl, bh;
#pragma unroll
    for (int h = 0; h < 8; ++h) {
      bl[h]     = (_Float16)fmaf(patch[h],      inv, bias);
      bl[8 + h] = (_Float16)fmaf(patch[16 + h], inv, bias);
      bh[h]     = (_Float16)fmaf(patch[8 + h],  inv, bias);
    }
    bh[8] = (_Float16)fmaf(patch[24], inv, bias);
#pragma unroll
    for (int h = 9; h < 16; ++h) bh[h] = (_Float16)0.f;

    const v8i bli = __builtin_bit_cast(v8i, bl);
    const v8i bhi = __builtin_bit_cast(v8i, bh);
    v8i blx, bhx;
#pragma unroll
    for (int t = 0; t < 8; ++t) blx[t] = __shfl_xor(bli[t], 16, 32);
#pragma unroll
    for (int t = 0; t < 5; ++t) bhx[t] = __shfl_xor(bhi[t], 16, 32);
#pragma unroll
    for (int t = 5; t < 8; ++t) bhx[t] = 0;

    // two N-tiles for this e: f-columns 0-15 and 16-31
#pragma unroll
    for (int sub = 0; sub < 2; ++sub) {
      v8i fi;
#pragma unroll
      for (int t = 0; t < 8; ++t)
        fi[t] = sub ? (half ? bhi[t] : blx[t])    // cols f=16-31
                    : (half ? bhx[t] : bli[t]);   // cols f=0-15
      const v16h bf = __builtin_bit_cast(v16h, fi);

      const int nbase = obase + e * WW + sub * 16 + l;
#pragma unroll
      for (int mt = 0; mt < 2; ++mt) {
        v8f c = {};
        c = __builtin_amdgcn_wmma_f32_16x16x32_f16(
                /*neg_a=*/false, afrag[mt], /*neg_b=*/false, bf,
                /*c_mod=*/(short)0, c, /*reuse_a=*/false, /*reuse_b=*/false);
        // D layout: lanes 0-15 VGPR j -> M=j ; lanes 16-31 VGPR j -> M=j+8
#pragma unroll
        for (int j = 0; j < 8; ++j) {
          const int w_out = mt * 16 + half * 8 + j;
          // streaming output (335 MB, never re-read): non-temporal store
          __builtin_nontemporal_store(c[j], &out[nbase + w_out * WSTRIDE]);
        }
      }
    }
  }
}

extern "C" void kernel_launch(void* const* d_in, const int* in_sizes, int n_in,
                              void* d_out, int out_size, void* d_ws, size_t ws_size,
                              hipStream_t stream) {
  const float* x1 = (const float*)d_in[0];
  const float* x2 = (const float*)d_in[1];
  float* out = (float*)d_out;
  // one block per (b, r, 8-channel group): 8*32*8 = 2048 blocks of 256 threads
  const int nblocks = BB * HH * (CC / 8);
  hipLaunchKernelGGL(ncl_wmma_kernel, dim3(nblocks), dim3(256), 0, stream,
                     x1, x2, out);
}
